// DetectionHead_40613210751460
// MI455X (gfx1250) — compile-verified, matrix operand order checked by reading, not run on hardware
//
#include <hip/hip_runtime.h>
#include <hip/hip_bf16.h>
#include <math.h>

#define NPTS   4096
#define C      128
#define M_CL   128
#define R2     0.25f
#define NMS_THR 0.25f
#define PITCH  136   // halves; rows 272B apart -> every 8-half chunk is 16B aligned
#define TP     32    // points per tile in cluster kernel

typedef _Float16 v16h __attribute__((ext_vector_type(16)));
typedef _Float16 h8   __attribute__((ext_vector_type(8)));
typedef float    v8f  __attribute__((ext_vector_type(8)));
typedef unsigned int u32x4 __attribute__((ext_vector_type(4)));
typedef int          i32x4 __attribute__((ext_vector_type(4)));
typedef int          i32x8 __attribute__((ext_vector_type(8)));

#if defined(__gfx1250__) && __has_builtin(__builtin_amdgcn_tensor_load_to_lds) && \
    __has_builtin(__builtin_amdgcn_s_wait_tensorcnt)
#define USE_TDM 1
#else
#define USE_TDM 0
#endif

// Load a 16x32 f16 WMMA fragment (A-style; B uses transposed LDS with row==N).
// Lane L<16 : row = base_row+L,    K = k0+{0..7} and k0+{16..23}
// Lane L>=16: row = base_row+L-16, K = k0+{8..15} and k0+{24..31}
__device__ __forceinline__ v16h frag_ld(const _Float16* base, int row, int k0, int lane) {
    int r  = row + (lane & 15);
    int kb = k0 + ((lane & 16) ? 8 : 0);
    const _Float16* p = base + r * PITCH + kb;
    h8 lo = *reinterpret_cast<const h8*>(p);
    h8 hi = *reinterpret_cast<const h8*>(p + 16);
    return __builtin_shufflevector(lo, hi, 0,1,2,3,4,5,6,7,8,9,10,11,12,13,14,15);
}

__device__ __forceinline__ v8f wmma16(v16h a, v16h b, v8f c) {
    return __builtin_amdgcn_wmma_f32_16x16x32_f16(false, a, false, b, (short)0, c, false, false);
}

#if USE_TDM
// TDM D# for a 2-D f16 tile load: tile_dim0=128 (row of channels), tile_dim1=TP rows,
// tensor_dim0=128, tensor_dim1=NPTS, tensor_dim0_stride=128 elements (data_size=2B).
__device__ __forceinline__ void tdm_load_f1_tile(const _Float16* gsrc, unsigned lds_off) {
    unsigned long long ga = (unsigned long long)(uintptr_t)gsrc;
    u32x4 g0;
    g0[0] = 1u;                                   // count=1 (valid), user mode
    g0[1] = lds_off;                              // lds_addr (bytes)
    g0[2] = (unsigned)(ga & 0xffffffffu);         // global_addr[31:0]
    g0[3] = (unsigned)(ga >> 32) | 0x80000000u;   // global_addr[56:32] | type=2<<30
    i32x8 g1;
    g1[0] = 1 << 16;                              // data_size=1 -> 2 bytes
    g1[1] = (int)(128u << 16);                    // tensor_dim0 = 128 (bits[79:48])
    g1[2] = (int)(4096u << 16);                   // tensor_dim1 = 4096 (bits[111:80])
    g1[3] = (int)(128u << 16);                    // tile_dim0 = 128 (bits[127:112])
    g1[4] = TP;                                   // tile_dim1 = TP, tile_dim2 = 0
    g1[5] = 128;                                  // tensor_dim0_stride = 128
    g1[6] = 0;
    g1[7] = 0;
    i32x4 z4 = {0, 0, 0, 0};
    i32x8 z8 = {0, 0, 0, 0, 0, 0, 0, 0};
    // clang-23 / therock lane: 6-arg form (g0, g1, g2, g3, extra, cpol)
    __builtin_amdgcn_tensor_load_to_lds(g0, g1, z4, z4, z8, 0);
}
#endif

// ---------------- Kernel A: farthest point sampling (1 block) ----------------
__global__ void __launch_bounds__(256) fps_kernel(const float* __restrict__ pts,
                                                  float* __restrict__ centers) {
    __shared__ float px[NPTS], py[NPTS], pz[NPTS];
    __shared__ float rv[256];
    __shared__ int   ri[256];
    __shared__ float csel[3];
    int tid = threadIdx.x;
    for (int i = tid; i < NPTS; i += 256) {
        px[i] = pts[i*3]; py[i] = pts[i*3+1]; pz[i] = pts[i*3+2];
    }
    __syncthreads();
    float md[16];
    float cx = px[0], cy = py[0], cz = pz[0];
    for (int k = 0; k < 16; ++k) {
        int p = tid*16 + k;
        float dx = px[p]-cx, dy = py[p]-cy, dz = pz[p]-cz;
        md[k] = dx*dx + dy*dy + dz*dz;
    }
    if (tid == 0) { centers[0] = cx; centers[1] = cy; centers[2] = cz; }
    for (int it = 1; it < M_CL; ++it) {
        float bv = -1.0f; int bi = 0;
        for (int k = 0; k < 16; ++k) {          // '>' keeps first (lowest) index on ties
            int p = tid*16 + k;
            if (md[k] > bv) { bv = md[k]; bi = p; }
        }
        rv[tid] = bv; ri[tid] = bi;
        __syncthreads();
        for (int s = 128; s > 0; s >>= 1) {
            if (tid < s) {
                float ov = rv[tid+s]; int oi = ri[tid+s];
                if (ov > rv[tid] || (ov == rv[tid] && oi < ri[tid])) { rv[tid] = ov; ri[tid] = oi; }
            }
            __syncthreads();
        }
        if (tid == 0) {
            int b = ri[0];
            csel[0] = px[b]; csel[1] = py[b]; csel[2] = pz[b];
            centers[it*3] = px[b]; centers[it*3+1] = py[b]; centers[it*3+2] = pz[b];
        }
        __syncthreads();
        float sx = csel[0], sy = csel[1], sz = csel[2];
        for (int k = 0; k < 16; ++k) {
            int p = tid*16 + k;
            float dx = px[p]-sx, dy = py[p]-sy, dz = pz[p]-sz;
            md[k] = fminf(md[k], dx*dx + dy*dy + dz*dz);
        }
        __syncthreads();
    }
}

// ---------------- Kernel B: F1 = feat @ W1[3:]  -> f16 [4096,128] ----------------
__global__ void __launch_bounds__(256) feat_gemm_kernel(const float* __restrict__ feat,
                                                        const float* __restrict__ W1,
                                                        _Float16* __restrict__ F1h) {
    __shared__ __align__(16) _Float16 aT[64 * PITCH];
    __shared__ __align__(16) _Float16 bT[128 * PITCH];
    int tid = threadIdx.x, lane = tid & 31, w = tid >> 5;
    int r0 = blockIdx.x * 64;
    for (int i = tid; i < 64*C; i += 256) {
        int r = i >> 7, c = i & 127;
        aT[r*PITCH + c] = (_Float16)feat[(r0 + r)*C + c];
    }
    for (int i = tid; i < C*C; i += 256) {
        int k = i >> 7, n = i & 127;
        bT[n*PITCH + k] = (_Float16)W1[(3 + k)*C + n];
    }
    __syncthreads();
    int rb = (w & 3) * 16;
    int ch = (w >> 2) * 64;
    for (int j = 0; j < 4; ++j) {
        int n0 = ch + j*16;
        v8f acc = {};
        for (int k0 = 0; k0 < 128; k0 += 32)
            acc = wmma16(frag_ld(aT, rb, k0, lane), frag_ld(bT, n0, k0, lane), acc);
        int n = n0 + (lane & 15);
        int rbase = rb + ((lane & 16) ? 8 : 0);
        for (int e = 0; e < 8; ++e)
            F1h[(r0 + rbase + e)*C + n] = (_Float16)acc[e];
    }
}

// ---------------- Kernel C: fused grouping MLP + masked max pool (1 block/cluster) ----
// F1 tiles stream into an LDS double-buffer via the Tensor Data Mover (ASYNC path);
// h1 = relu(F1 + rel@W1[:3] + b1) built in LDS; h2 = h1@W2 via chained WMMA; masked
// channel-max folded straight out of the accumulator fragments.
__global__ void __launch_bounds__(256) cluster_kernel(const float* __restrict__ pts,
                                                      const float* __restrict__ W1,
                                                      const float* __restrict__ b1,
                                                      const float* __restrict__ W2,
                                                      const float* __restrict__ b2,
                                                      const _Float16* __restrict__ F1h,
                                                      const float* __restrict__ centers,
                                                      float* __restrict__ G) {
    __shared__ __align__(16) _Float16 w2t[128 * PITCH];  // W2 transposed [n][k]  (34816 B)
    __shared__ __align__(16) _Float16 hT[TP * PITCH];    // h1 tile [p][c]        (8704 B)
    __shared__ __align__(16) _Float16 f1s[2][TP * C];    // F1 staging x2         (16384 B)
    __shared__ float relx[TP], rely[TP], relz[TP], d2s[TP];
    __shared__ float w1x[C], w1y[C], w1z[C], b1s[C], b2s[C];
    __shared__ float gmax[C];
    __shared__ float part[2 * C];
    int tid = threadIdx.x, lane = tid & 31, w = tid >> 5;
    int m = blockIdx.x;
    float cx = centers[m*3], cy = centers[m*3+1], cz = centers[m*3+2];
    for (int i = tid; i < C*C; i += 256) {
        int k = i >> 7, n = i & 127;
        w2t[n*PITCH + k] = (_Float16)W2[i];
    }
    for (int i = tid; i < C; i += 256) {
        w1x[i] = W1[i]; w1y[i] = W1[C + i]; w1z[i] = W1[2*C + i];
        b1s[i] = b1[i]; b2s[i] = b2[i];
        gmax[i] = -INFINITY;
    }
    __syncthreads();
    int rb = (w & 1) * 16;           // 2 row blocks of 16 points
    int ch = (w >> 1) * 32;          // 4 column quarters of 32 channels
#if USE_TDM
    if (tid == 0)
        tdm_load_f1_tile(F1h, (unsigned)(uintptr_t)&f1s[0][0]);
#endif
    for (int t = 0; t < NPTS/TP; ++t) {
        int p0 = t * TP;
        int buf = t & 1;
#if USE_TDM
        if (tid == 0) __builtin_amdgcn_s_wait_tensorcnt(0);  // tile t landed in f1s[buf]
#endif
        if (tid < TP) {
            int p = p0 + tid;
            float x = pts[p*3], y = pts[p*3+1], z = pts[p*3+2];
            relx[tid] = cx - x*2.0f;  // center - p/RADIUS, RADIUS=0.5
            rely[tid] = cy - y*2.0f;
            relz[tid] = cz - z*2.0f;
            float dx = cx - x, dy = cy - y, dz = cz - z;
            d2s[tid] = dx*dx + dy*dy + dz*dz;
        }
        __syncthreads();   // publishes f1s[buf] (TDM) + rel/d2
#if USE_TDM
        if (tid == 0 && t + 1 < NPTS/TP)   // overlap tile t+1 DMA with tile t compute
            tdm_load_f1_tile(F1h + (size_t)(p0 + TP)*C, (unsigned)(uintptr_t)&f1s[buf ^ 1][0]);
#else
        for (int i = tid; i < (TP*C)/8; i += 256) {
            int p = i >> 4, c0 = (i & 15) * 8;
            *reinterpret_cast<h8*>(&f1s[buf][p*C + c0]) =
                *reinterpret_cast<const h8*>(&F1h[(size_t)(p0 + p)*C + c0]);
        }
        if (tid < 128 && t + 1 < NPTS/TP)
            __builtin_prefetch(&F1h[(size_t)(p0 + TP)*C + tid*64], 0, 1);
        __syncthreads();
#endif
        // h1 = relu(F1 + rel @ W1[:3] + b1), 16B-vectorized
        for (int i = tid; i < (TP*C)/8; i += 256) {
            int p = i >> 4, c0 = (i & 15) * 8;
            h8 f = *reinterpret_cast<const h8*>(&f1s[buf][p*C + c0]);
            float rx = relx[p], ry = rely[p], rz = relz[p];
            h8 o;
            for (int e = 0; e < 8; ++e) {
                int c = c0 + e;
                float v = (float)f[e] + b1s[c] + rx*w1x[c] + ry*w1y[c] + rz*w1z[c];
                o[e] = (_Float16)fmaxf(v, 0.0f);
            }
            *reinterpret_cast<h8*>(&hT[p*PITCH + c0]) = o;
        }
        __syncthreads();
        for (int j = 0; j < 2; ++j) {
            int n0 = ch + j*16;
            v8f acc = {};
            for (int k0 = 0; k0 < 128; k0 += 32)
                acc = wmma16(frag_ld(hT, rb, k0, lane), frag_ld(w2t, n0, k0, lane), acc);
            int n = n0 + (lane & 15);
            int rbase = rb + ((lane & 16) ? 8 : 0);
            float bb = b2s[n];
            float pm = -INFINITY;
            for (int e = 0; e < 8; ++e) {
                float v = fmaxf(acc[e] + bb, 0.0f);
                v = (d2s[rbase + e] < R2) ? v : -INFINITY;
                pm = fmaxf(pm, v);
            }
            pm = fmaxf(pm, __shfl_xor(pm, 16, 32)); // lanes l and l+16 share channel n
            if (lane < 16) part[(w & 1)*C + n] = pm;
        }
        __syncthreads();
        if (tid < C)
            gmax[tid] = fmaxf(gmax[tid], fmaxf(part[tid], part[C + tid]));
        __syncthreads();
    }
#if USE_TDM
    if (tid == 0) __builtin_amdgcn_s_wait_tensorcnt(0);
#endif
    if (tid < C) G[m*C + tid] = gmax[tid];
}

// ---------------- Kernel D: head MLP relu(relu(G@W3+b3)@W4+b4)@Wf+bf -> sigmoid ----------------
__global__ void __launch_bounds__(256) head_kernel(const float* __restrict__ G,
                                                   const float* __restrict__ W3,
                                                   const float* __restrict__ b3,
                                                   const float* __restrict__ W4,
                                                   const float* __restrict__ b4,
                                                   const float* __restrict__ Wf,
                                                   const float* __restrict__ bfv,
                                                   float* __restrict__ sc_out,
                                                   float* __restrict__ co_out) {
    __shared__ __align__(16) _Float16 abuf[64 * PITCH];
    __shared__ __align__(16) _Float16 bbuf[128 * PITCH];
    __shared__ float bias[C];
    int tid = threadIdx.x, lane = tid & 31, w = tid >> 5;
    int r0 = blockIdx.x * 64;
    for (int i = tid; i < 64*C; i += 256) {
        int r = i >> 7, c = i & 127;
        abuf[r*PITCH + c] = (_Float16)G[(r0 + r)*C + c];
    }
    for (int i = tid; i < C*C; i += 256) {
        int k = i >> 7, n = i & 127;
        bbuf[n*PITCH + k] = (_Float16)W3[i];
    }
    for (int i = tid; i < C; i += 256) bias[i] = b3[i];
    __syncthreads();
    int rb = (w & 3) * 16, ch = (w >> 2) * 64;
    v8f accs[4];
    for (int j = 0; j < 4; ++j) {
        int n0 = ch + j*16;
        v8f acc = {};
        for (int k0 = 0; k0 < 128; k0 += 32)
            acc = wmma16(frag_ld(abuf, rb, k0, lane), frag_ld(bbuf, n0, k0, lane), acc);
        accs[j] = acc;
    }
    __syncthreads();
    for (int j = 0; j < 4; ++j) {      // H3 = relu(acc + b3) -> abuf
        int n0 = ch + j*16;
        int n = n0 + (lane & 15);
        int rbase = rb + ((lane & 16) ? 8 : 0);
        float bb = bias[n];
        for (int e = 0; e < 8; ++e)
            abuf[(rbase + e)*PITCH + n] = (_Float16)fmaxf(accs[j][e] + bb, 0.0f);
    }
    __syncthreads();
    for (int i = tid; i < C*C; i += 256) {
        int k = i >> 7, n = i & 127;
        bbuf[n*PITCH + k] = (_Float16)W4[i];
    }
    for (int i = tid; i < C; i += 256) bias[i] = b4[i];
    __syncthreads();
    for (int j = 0; j < 4; ++j) {
        int n0 = ch + j*16;
        v8f acc = {};
        for (int k0 = 0; k0 < 128; k0 += 32)
            acc = wmma16(frag_ld(abuf, rb, k0, lane), frag_ld(bbuf, n0, k0, lane), acc);
        accs[j] = acc;
    }
    __syncthreads();
    for (int j = 0; j < 4; ++j) {      // H4 = relu(acc + b4) -> abuf
        int n0 = ch + j*16;
        int n = n0 + (lane & 15);
        int rbase = rb + ((lane & 16) ? 8 : 0);
        float bb = bias[n];
        for (int e = 0; e < 8; ++e)
            abuf[(rbase + e)*PITCH + n] = (_Float16)fmaxf(accs[j][e] + bb, 0.0f);
    }
    __syncthreads();
    for (int o = tid; o < 64*7; o += 256) {   // boxes = H4 @ Wf + bf, then sigmoid
        int r = o / 7, jj = o % 7;
        float s = bfv[jj];
        for (int k = 0; k < C; ++k) s += (float)abuf[r*PITCH + k] * Wf[k*7 + jj];
        float sg = 1.0f / (1.0f + __expf(-s));
        int row = r0 + r;
        if (jj == 0) sc_out[row] = sg;
        else         co_out[row*6 + (jj - 1)] = sg;
    }
}

// ---------------- Kernel E: 3D axis-aligned IoU NMS over 128 boxes ----------------
__global__ void __launch_bounds__(128) nms_kernel(const float* __restrict__ sc,
                                                  const float* __restrict__ co,
                                                  float* __restrict__ out) {
    __shared__ float s[128];
    __shared__ float blo[128][3], bhi[128][3], vol[128];
    __shared__ int order[128];
    __shared__ int rank[128];
    __shared__ int supp[128];
    int i = threadIdx.x;
    float sv = sc[i];
    s[i] = sv;
    float c0 = co[i*6+0], c1 = co[i*6+1], c2 = co[i*6+2];
    float d0 = co[i*6+3], d1 = co[i*6+4], d2 = co[i*6+5];
    blo[i][0] = c0 - 0.5f*d0; blo[i][1] = c1 - 0.5f*d1; blo[i][2] = c2 - 0.5f*d2;
    bhi[i][0] = c0 + 0.5f*d0; bhi[i][1] = c1 + 0.5f*d1; bhi[i][2] = c2 + 0.5f*d2;
    vol[i] = d0 * d1 * d2;
    supp[i] = 0;
    __syncthreads();
    int r = 0;                         // stable descending rank == argsort(-scores)
    for (int j = 0; j < 128; ++j) {
        float sj = s[j];
        if (sj > sv || (sj == sv && j < i)) r++;
    }
    rank[i] = r;
    order[r] = i;
    __syncthreads();
    for (int pos = 0; pos < 127; ++pos) {
        int idx = order[pos];
        if (!supp[idx] && rank[i] > pos && !supp[i]) {
            float inter = 1.0f;
            for (int d = 0; d < 3; ++d) {
                float l = fmaxf(blo[i][d], blo[idx][d]);
                float h = fminf(bhi[i][d], bhi[idx][d]);
                inter *= fmaxf(h - l, 0.0f);
            }
            float iou = inter / (vol[i] + vol[idx] - inter + 1e-8f);
            if (iou > NMS_THR) supp[i] = 1;
        }
        __syncthreads();
    }
    float keep = supp[i] ? 0.0f : 1.0f;
    for (int d = 0; d < 6; ++d) out[i*6 + d] = co[i*6 + d] * keep;
}

extern "C" void kernel_launch(void* const* d_in, const int* in_sizes, int n_in,
                              void* d_out, int out_size, void* d_ws, size_t ws_size,
                              hipStream_t stream) {
    (void)in_sizes; (void)n_in; (void)out_size; (void)ws_size;
    const float* pts  = (const float*)d_in[0];   // [4096,3]
    const float* feat = (const float*)d_in[1];   // [4096,128]
    const float* W1   = (const float*)d_in[2];   // [131,128]
    const float* b1   = (const float*)d_in[3];
    const float* W2   = (const float*)d_in[4];   // [128,128]
    const float* b2   = (const float*)d_in[5];
    const float* W3   = (const float*)d_in[6];
    const float* b3   = (const float*)d_in[7];
    const float* W4   = (const float*)d_in[8];
    const float* b4   = (const float*)d_in[9];
    const float* Wf   = (const float*)d_in[10];  // [128,7]
    const float* bfv  = (const float*)d_in[11];

    char* ws = (char*)d_ws;
    float*    centers = (float*)ws;                              // 128*3 f32
    _Float16* F1h     = (_Float16*)(ws + 2048);                  // 4096*128 f16 = 1MB
    float*    G       = (float*)(ws + 2048 + 1048576);           // 128*128 f32
    float*    sc      = (float*)(ws + 2048 + 1048576 + 65536);   // 128 f32
    float*    co      = sc + 128;                                // 128*6 f32

    fps_kernel      <<<1,   256, 0, stream>>>(pts, centers);
    feat_gemm_kernel<<<64,  256, 0, stream>>>(feat, W1, F1h);
    cluster_kernel  <<<128, 256, 0, stream>>>(pts, W1, b1, W2, b2, F1h, centers, G);
    head_kernel     <<<2,   256, 0, stream>>>(G, W3, b3, W4, b4, Wf, bfv, sc, co);
    nms_kernel      <<<1,   128, 0, stream>>>(sc, co, (float*)d_out);
}
